// gatedRNNEncoder1_80985903334240
// MI455X (gfx1250) — matrix operand order repaired and checked by reading, not compiled, not run on hardware
//
#include <hip/hip_runtime.h>
#include <cmath>

// ---------------------------------------------------------------------------
// gatedRNNEncoder1 for MI455X (gfx1250, wave32, WMMA) — persistent-kernel form.
//
// One persistent kernel (32 blocks x 256 threads) runs all 400 steps:
//   phase 1: gates = [vJ | c_t] @ Wc^T via v_wmma_f32_16x16x32_bf16
//            (B operand resident in LDS for the whole run) + attention -> cT
//   grid barrier (32 arrivals, agent-scope fences)
//   phase 2: LSTM cell -> h_t -> V[:,t,:], vJ, and next A = [h_t | c_{t+1}]
//   grid barrier
//
// Workspace layout (bytes), ~6.3 MB total:
//   Abuf  bf16 [128][1024]   @ 0
//   Wc    bf16 [2048][1024]  @ 262144
//   gates f32  [128][2048]   @ 4456448
//   cT    f32  [128][512]    @ 5505024
//   vJ    f32  [128][512]    @ 5767168
//   qp    f32  [128][48]     @ 6029312
//   cpass f32  [128][400]    @ 6053888
//   barcnt u32               @ 6258688
// ---------------------------------------------------------------------------

typedef __bf16 bf16_t;
typedef __attribute__((ext_vector_type(16))) __bf16 v16bf;
typedef __attribute__((ext_vector_type(8)))  __bf16 v8bf;
typedef __attribute__((ext_vector_type(8)))  float  v8f;

static constexpr int Bb   = 128;
static constexpr int LC   = 400;
static constexpr int LQ   = 48;
static constexpr int H    = 512;
static constexpr int N4H  = 2048;   // 4*H
static constexpr int KC   = 1024;   // combined K = 2*H
static constexpr int NBLK = 32;
static constexpr int NTHR = 256;
static constexpr int BSTRH = 1032;  // LDS B row stride in halves (2064 B, bank-skewed)

__device__ __forceinline__ float sigf(float x) { return 1.0f / (1.0f + __expf(-x)); }

// ---------------- precompute kernels ----------------

__global__ __launch_bounds__(256) void pack_wc(const float* __restrict__ W_ih,
                                               const float* __restrict__ W_hh,
                                               bf16_t* __restrict__ Wc) {
  int idx = blockIdx.x * 256 + threadIdx.x;        // 2048*1024
  int n = idx >> 10, k = idx & 1023;
  float v = (k < H) ? W_ih[n * H + k] : W_hh[n * H + (k - H)];
  Wc[idx] = (bf16_t)v;
}

__global__ __launch_bounds__(256) void calc_qp(const float* __restrict__ q,
                                               const float* __restrict__ w_quest,
                                               float* __restrict__ qp) {
  int idx = blockIdx.x * 256 + threadIdx.x;        // B*LQ = 6144
  if (idx >= Bb * LQ) return;
  const float* row = q + (size_t)idx * H;
  float s = 0.f;
  for (int h = 0; h < H; ++h) s += row[h] * w_quest[h];
  qp[idx] = s;
}

__global__ __launch_bounds__(256) void calc_cpass(const float* __restrict__ c,
                                                  const float* __restrict__ w_pass,
                                                  float* __restrict__ cpass) {
  int idx = blockIdx.x * 256 + threadIdx.x;        // B*LC = 51200
  if (idx >= Bb * LC) return;
  const float* row = c + (size_t)idx * H;
  float s = 0.f;
  for (int h = 0; h < H; ++h) s += row[h] * w_pass[h];
  cpass[idx] = s;
}

__global__ __launch_bounds__(256) void init_state(const float* __restrict__ c,
                                                  float* __restrict__ vJ,
                                                  bf16_t* __restrict__ Abuf,
                                                  unsigned* __restrict__ barcnt) {
  int idx = blockIdx.x * 256 + threadIdx.x;        // B*H = 65536
  if (idx == 0) *barcnt = 0u;
  int b = idx >> 9, h = idx & 511;
  vJ[idx] = 0.f;
  Abuf[b * KC + h] = (bf16_t)0.f;
  Abuf[b * KC + H + h] = (bf16_t)c[((size_t)b * LC) * H + h];
}

// ---------------- grid barrier (32 blocks) ----------------
__device__ __forceinline__ void grid_barrier(unsigned* cnt, unsigned target) {
  __builtin_amdgcn_fence(__ATOMIC_RELEASE, "agent");   // drain this thread's stores
  __syncthreads();
  if (threadIdx.x == 0) {
    __hip_atomic_fetch_add(cnt, 1u, __ATOMIC_RELAXED, __HIP_MEMORY_SCOPE_AGENT);
    while (__hip_atomic_load(cnt, __ATOMIC_RELAXED, __HIP_MEMORY_SCOPE_AGENT) < target)
      __builtin_amdgcn_s_sleep(1);
  }
  __syncthreads();
  __builtin_amdgcn_fence(__ATOMIC_ACQUIRE, "agent");   // invalidate stale near caches
}

// ---------------- persistent kernel: all 400 steps ----------------
__global__ __launch_bounds__(NTHR)
void rnn_persistent(const bf16_t* __restrict__ Wc,
                    bf16_t* __restrict__ Abuf,
                    float* __restrict__ gates,
                    float* __restrict__ cT,
                    float* __restrict__ vJ,
                    const float* __restrict__ qp,
                    const float* __restrict__ cpass,
                    const float* __restrict__ q,
                    const float* __restrict__ w_prev,
                    const float* __restrict__ b_ih,
                    const float* __restrict__ b_hh,
                    const float* __restrict__ c,
                    float* __restrict__ V,
                    unsigned* __restrict__ barcnt) {
  __shared__ bf16_t Bsh[64 * BSTRH];   // 132096 B: this block's weight stripe, whole run
  __shared__ float  red[NTHR];
  __shared__ float  ash[64];

  const int tid = threadIdx.x;
  const int n0  = blockIdx.x * 64;     // N stripe [n0, n0+64)

  // one-time: stage Wc stripe (64 cols x 1024 K bf16) into LDS, 16B chunks
  for (int i = tid; i < 64 * 128; i += NTHR) {
    int row = i >> 7, seg = i & 127;
    *(uint4*)((char*)Bsh + row * (BSTRH * 2) + seg * 16) =
        *(const uint4*)((const char*)(Wc + (size_t)(n0 + row) * KC) + seg * 16);
  }
  __syncthreads();

  const int wave = tid >> 5, lane = tid & 31;
  const int laneHi = lane >> 4, l15 = lane & 15;
  const int m0 = wave * 16;            // 8 waves cover M = 0..127

  const bf16_t* Arow = Abuf + (size_t)(m0 + l15) * KC;
  const bf16_t* Bl0 = Bsh + ( 0 + l15) * BSTRH + laneHi * 8;
  const bf16_t* Bl1 = Bsh + (16 + l15) * BSTRH + laneHi * 8;
  const bf16_t* Bl2 = Bsh + (32 + l15) * BSTRH + laneHi * 8;
  const bf16_t* Bl3 = Bsh + (48 + l15) * BSTRH + laneHi * 8;

  unsigned ep = 0;

  for (int t = 0; t < LC; ++t) {
    // ---------- phase 1a: WMMA GEMM: gates = A @ Wc^T (M=128,N=2048,K=1024) ----------
    v8f acc0 = {}, acc1 = {}, acc2 = {}, acc3 = {};
    for (int kb = 0; kb < KC / 32; ++kb) {
      const int kbase = kb * 32;
      v8bf alo = *(const v8bf*)(Arow + kbase + laneHi * 8);
      v8bf ahi = *(const v8bf*)(Arow + kbase + 16 + laneHi * 8);
      v16bf af;
#pragma unroll
      for (int e = 0; e < 8; ++e) { af[e] = alo[e]; af[e + 8] = ahi[e]; }

      v8bf p0 = *(const v8bf*)(Bl0 + kbase), q0 = *(const v8bf*)(Bl0 + kbase + 16);
      v8bf p1 = *(const v8bf*)(Bl1 + kbase), q1 = *(const v8bf*)(Bl1 + kbase + 16);
      v8bf p2 = *(const v8bf*)(Bl2 + kbase), q2 = *(const v8bf*)(Bl2 + kbase + 16);
      v8bf p3 = *(const v8bf*)(Bl3 + kbase), q3 = *(const v8bf*)(Bl3 + kbase + 16);
      v16bf b0, b1, b2, b3;
#pragma unroll
      for (int e = 0; e < 8; ++e) {
        b0[e] = p0[e]; b0[e + 8] = q0[e];
        b1[e] = p1[e]; b1[e + 8] = q1[e];
        b2[e] = p2[e]; b2[e + 8] = q2[e];
        b3[e] = p3[e]; b3[e + 8] = q3[e];
      }
      acc0 = __builtin_amdgcn_wmma_f32_16x16x32_bf16(false, af, false, b0, (short)0, acc0, false, false);
      acc1 = __builtin_amdgcn_wmma_f32_16x16x32_bf16(false, af, false, b1, (short)0, acc1, false, false);
      acc2 = __builtin_amdgcn_wmma_f32_16x16x32_bf16(false, af, false, b2, (short)0, acc2, false, false);
      acc3 = __builtin_amdgcn_wmma_f32_16x16x32_bf16(false, af, false, b3, (short)0, acc3, false, false);
    }
#pragma unroll
    for (int r = 0; r < 8; ++r) {
      int row = m0 + laneHi * 8 + r;
      float* g = gates + (size_t)row * N4H + (n0 + l15);
      g[0]  = acc0[r];
      g[16] = acc1[r];
      g[32] = acc2[r];
      g[48] = acc3[r];
    }

    // ---------- phase 1b: attention, 4 batch rows per block ----------
    for (int r = 0; r < 4; ++r) {
      const int b = blockIdx.x * 4 + r;
      float p = 0.f;
      for (int h = tid; h < H; h += NTHR) p += vJ[b * H + h] * w_prev[h];
      red[tid] = p;
      __syncthreads();
      for (int s = 128; s > 0; s >>= 1) {
        if (tid < s) red[tid] += red[tid + s];
        __syncthreads();
      }
      const float dp = red[0] + cpass[b * LC + t];
      if (tid < LQ) ash[tid] = tanhf(qp[b * LQ + tid] + dp);
      __syncthreads();
      if (tid == 0) {
        float mx = ash[0];
        for (int l = 1; l < LQ; ++l) mx = fmaxf(mx, ash[l]);
        float sum = 0.f;
        for (int l = 0; l < LQ; ++l) { float e = __expf(ash[l] - mx); ash[l] = e; sum += e; }
        ash[48] = 1.0f / sum;
      }
      __syncthreads();
      const float inv = ash[48];
      const float* qb = q + (size_t)b * LQ * H;
      for (int h = tid; h < H; h += NTHR) {
        float a = 0.f;
#pragma unroll 8
        for (int l = 0; l < LQ; ++l) a += ash[l] * qb[l * H + h];
        cT[b * H + h] = a * inv;
      }
      __syncthreads();
    }

    grid_barrier(barcnt, (++ep) * NBLK);

    // ---------- phase 2: LSTM cell, strided over B*H ----------
    for (int i = blockIdx.x * NTHR + tid; i < Bb * H; i += NBLK * NTHR) {
      int b = i >> 9, h = i & 511;
      const float* g = gates + (size_t)b * N4H;
      float ig = sigf (g[h]         + b_ih[h]         + b_hh[h]);
      float fg = sigf (g[H + h]     + b_ih[H + h]     + b_hh[H + h]);
      float gg = tanhf(g[2 * H + h] + b_ih[2 * H + h] + b_hh[2 * H + h]);
      float og = sigf (g[3 * H + h] + b_ih[3 * H + h] + b_hh[3 * H + h]);
      float cn = fg * cT[b * H + h] + ig * gg;
      float hv = og * tanhf(cn);
      V[((size_t)b * LC + t) * H + h] = hv;
      vJ[b * H + h] = hv;
      Abuf[b * KC + h] = (bf16_t)hv;
      if (t + 1 < LC) Abuf[b * KC + H + h] = (bf16_t)c[((size_t)b * LC + (t + 1)) * H + h];
    }

    grid_barrier(barcnt, (++ep) * NBLK);
  }
}

// ---------------- launch ----------------
extern "C" void kernel_launch(void* const* d_in, const int* in_sizes, int n_in,
                              void* d_out, int out_size, void* d_ws, size_t ws_size,
                              hipStream_t stream) {
  const float* c       = (const float*)d_in[0];
  const float* q       = (const float*)d_in[1];
  // d_in[2], d_in[3]: masks (all-ones; unused by the reference computation)
  const float* w_quest = (const float*)d_in[4];
  const float* w_pass  = (const float*)d_in[5];
  const float* w_prev  = (const float*)d_in[6];
  const float* W_ih    = (const float*)d_in[7];
  const float* W_hh    = (const float*)d_in[8];
  const float* b_ih    = (const float*)d_in[9];
  const float* b_hh    = (const float*)d_in[10];
  float* V = (float*)d_out;

  char* ws = (char*)d_ws;
  bf16_t*   Abuf   = (bf16_t*)(ws + 0);
  bf16_t*   Wc     = (bf16_t*)(ws + 262144);
  float*    gates  = (float*)(ws + 4456448);
  float*    cT     = (float*)(ws + 5505024);
  float*    vJ     = (float*)(ws + 5767168);
  float*    qp     = (float*)(ws + 6029312);
  float*    cpass  = (float*)(ws + 6053888);
  unsigned* barcnt = (unsigned*)(ws + 6258688);

  pack_wc   <<<(N4H * KC) / 256, 256, 0, stream>>>(W_ih, W_hh, Wc);
  calc_qp   <<<(Bb * LQ + 255) / 256, 256, 0, stream>>>(q, w_quest, qp);
  calc_cpass<<<(Bb * LC + 255) / 256, 256, 0, stream>>>(c, w_pass, cpass);
  init_state<<<(Bb * H) / 256, 256, 0, stream>>>(c, vJ, Abuf, barcnt);

  rnn_persistent<<<NBLK, NTHR, 0, stream>>>(Wc, Abuf, gates, cT, vJ, qp, cpass,
                                            q, w_prev, b_ih, b_hh, c, V, barcnt);
}
